// Functional_78426102825419
// MI455X (gfx1250) — matrix operand order repaired and checked by reading, not compiled, not run on hardware
//
#include <hip/hip_runtime.h>
#include <math.h>

#define NZV     262144
#define LOG2NZ  18
#define HALFNZ  (NZV / 2)
#define NZHV    131073
#define BV      8
#define HVC     64
#define NIRRED  14
#define DZC     0.1f
#define GCC     15.0f
#define TWO_PI  6.28318530717958647692f

typedef __attribute__((ext_vector_type(16))) _Float16 v16h;
typedef __attribute__((ext_vector_type(8)))  float    v8f;

// 1) Load real input n (B x NZ) into complex ping buffer (rows 0..7)
__global__ void __launch_bounds__(256) loadreal_kernel(const float* __restrict__ nin,
                                                       float2* __restrict__ dst) {
  int gid = blockIdx.x * 256 + threadIdx.x;
  if (gid < BV * NZV) dst[gid] = make_float2(nin[gid], 0.0f);
}

// 2) Stockham radix-2 FFT pass (autosort, DIF). One butterfly per thread.
//    pass i: nn = NZ >> i, s = 1 << i.
//    y[q+s*2p] = a+b ; y[q+s*(2p+1)] = (a-b)*w_p, w_p = exp(sign*i*2*pi*p/nn)
__global__ void __launch_bounds__(256) fft_pass_kernel(const float2* __restrict__ src,
                                                       float2* __restrict__ dst,
                                                       int ls, int nn, float sign, int total) {
  int tid = blockIdx.x * 256 + threadIdx.x;
  if (tid >= total) return;
  int row = tid >> (LOG2NZ - 1);
  int t   = tid & (HALFNZ - 1);
  int s   = 1 << ls;
  int m   = nn >> 1;
  int p   = t >> ls;
  int q   = t & (s - 1);

  const float2* x = src + (size_t)row * NZV;
  float2*       y = dst + (size_t)row * NZV;

  float2 a = x[q + s * p];
  float2 b = x[q + s * (p + m)];
  float sx = a.x + b.x, sy = a.y + b.y;
  float dx = a.x - b.x, dy = a.y - b.y;
  float ang = sign * TWO_PI * (float)p / (float)nn;
  float sn, cs;
  __sincosf(ang, &sn, &cs);
  y[q + s * (2 * p)]     = make_float2(sx, sy);
  y[q + s * (2 * p + 1)] = make_float2(dx * cs - dy * sn, dx * sn + dy * cs);
}

// 3) w-filter: per-frequency 1->64->64->8 CELU MLP + envelope/norm + fac.
__global__ void __launch_bounds__(256) wfilter_kernel(const float* __restrict__ wW0,
                                                      const float* __restrict__ wb0,
                                                      const float* __restrict__ wW1,
                                                      const float* __restrict__ wb1,
                                                      const float* __restrict__ wW2,
                                                      const float* __restrict__ wb2,
                                                      float2* __restrict__ wtbuf) {
  __shared__ float sW1[HVC * HVC];
  __shared__ float sW2[8 * HVC];
  __shared__ float sW0[HVC], sB0[HVC], sB1[HVC], sB2[8];
  for (int i = threadIdx.x; i < HVC * HVC; i += 256) sW1[i] = wW1[i];
  for (int i = threadIdx.x; i < 8 * HVC;   i += 256) sW2[i] = wW2[i];
  for (int i = threadIdx.x; i < HVC;       i += 256) { sW0[i] = wW0[i]; sB0[i] = wb0[i]; sB1[i] = wb1[i]; }
  if (threadIdx.x < 8) sB2[threadIdx.x] = wb2[threadIdx.x];
  __syncthreads();

  int k = blockIdx.x * 256 + threadIdx.x;
  if (k >= NZHV) return;

  float Gz = TWO_PI * (float)k / ((float)NZV * DZC);
  float r  = Gz / GCC;
  float g  = fminf(r * r, 1.0f);

  float h0[HVC];
#pragma unroll
  for (int j = 0; j < HVC; ++j) {
    float x = sW0[j] * g + sB0[j];
    h0[j] = x > 0.0f ? x : (__expf(x) - 1.0f);
  }
  float h1[HVC];
  for (int o = 0; o < HVC; ++o) {
    float acc = sB1[o];
#pragma unroll
    for (int j = 0; j < HVC; ++j) acc += sW1[o * HVC + j] * h0[j];
    h1[o] = acc > 0.0f ? acc : (__expf(acc) - 1.0f);
  }
  float env = (1.0f - g) * (1.0f - g);
  for (int c = 0; c < 8; ++c) {
    float acc = sB2[c];
#pragma unroll
    for (int j = 0; j < HVC; ++j) acc += sW2[c * HVC + j] * h1[j];
    int P  = c & 3;
    int GP = 2 * P + (c >= 4 ? 1 : 0);
    float prod = 1.0f;
    for (int o = 1; o <= 9; o += 2) prod *= (float)(2 * GP + o);
    float norm = sqrtf(prod / 384.0f);
    if (c >= 4) norm *= (1.0f / GCC);
    float gp = 1.0f;
    for (int t = 0; t < P; ++t) gp *= g;
    float val = acc * env * gp * norm;
    wtbuf[(size_t)c * NZHV + k] = (c < 4) ? make_float2(val, 0.0f)
                                          : make_float2(0.0f, val * Gz);
  }
}

// 4) Spectral multiply: prod[f*8+b][k] = Wfull[f][k] * n_tilde[b][k] / NZ
__global__ void __launch_bounds__(256) specmul_kernel(const float2* __restrict__ ntil,
                                                      const float2* __restrict__ wt,
                                                      float2* __restrict__ prod) {
  int gid = blockIdx.x * 256 + threadIdx.x;
  if (gid >= 64 * NZV) return;
  int k = gid & (NZV - 1);
  int rr = gid >> LOG2NZ;
  int f = rr >> 3, b = rr & 7;
  float2 nv = ntil[(size_t)b * NZV + k];
  float2 w;
  if (k <= HALFNZ) {
    w = wt[(size_t)f * NZHV + k];
  } else {
    w = wt[(size_t)f * NZHV + (NZV - k)];
    w.y = -w.y;
  }
  const float invN = 1.0f / (float)NZV;
  prod[gid] = make_float2((w.x * nv.x - w.y * nv.y) * invN,
                          (w.x * nv.y + w.y * nv.x) * invN);
}

// 5) Pack f-MLP weights into WMMA A fragments (16x32 f16, contiguous k order used by this code):
//    lane L: half = L>>4, m = tile*16 + (L&15);
//    element e: k = half*16 + e.
__global__ void __launch_bounds__(256) prep_weights_kernel(const float* __restrict__ fW0,
                                                           const float* __restrict__ fW1,
                                                           const float* __restrict__ fW2,
                                                           _Float16* __restrict__ aW0,
                                                           _Float16* __restrict__ aW1,
                                                           _Float16* __restrict__ aW2) {
  for (int idx = threadIdx.x; idx < 4 * 32 * 16; idx += 256) {
    int e = idx & 15, lane = (idx >> 4) & 31, t = idx >> 9;
    int hf = lane >> 4, ml = lane & 15;
    int k = hf * 16 + e;
    int m = t * 16 + ml;
    aW0[idx] = (_Float16)((k < NIRRED) ? fW0[m * NIRRED + k] : 0.0f);
  }
  for (int idx = threadIdx.x; idx < 2 * 4 * 32 * 16; idx += 256) {
    int e = idx & 15, lane = (idx >> 4) & 31, t = (idx >> 9) & 3, kc = idx >> 11;
    int hf = lane >> 4, ml = lane & 15;
    int k = kc * 32 + hf * 16 + e;
    int m = t * 16 + ml;
    aW1[idx] = (_Float16)fW1[m * HVC + k];
  }
  for (int idx = threadIdx.x; idx < 2 * 32 * 16; idx += 256) {
    int e = idx & 15, lane = (idx >> 4) & 31, kc = idx >> 9;
    int hf = lane >> 4, ml = lane & 15;
    int k = kc * 32 + hf * 16 + e;
    aW2[idx] = (_Float16)((ml < NIRRED) ? fW2[ml * HVC + k] : 0.0f);
  }
}

// 6) f-MLP over all (b, z) points with WMMA. One wave handles 16 points/tile.
__global__ void __launch_bounds__(256) fmlp_kernel(const float2* __restrict__ conv,
                                                   const _Float16* __restrict__ aW0,
                                                   const _Float16* __restrict__ aW1,
                                                   const _Float16* __restrict__ aW2,
                                                   const float* __restrict__ fb0,
                                                   const float* __restrict__ fb1,
                                                   const float* __restrict__ fb2,
                                                   float* __restrict__ partials) {
  __shared__ __align__(32) _Float16 hb[8][16 * 64];   // per-wave [n][k] staging
  __shared__ float sc[8][16 * 16];                    // per-wave scalars [n][c]
  __shared__ float sb0[HVC], sb1[HVC], sb2[16];
  for (int i = threadIdx.x; i < HVC; i += 256) { sb0[i] = fb0[i]; sb1[i] = fb1[i]; }
  if (threadIdx.x < 16) sb2[threadIdx.x] = (threadIdx.x < NIRRED) ? fb2[threadIdx.x] : 0.0f;
  __syncthreads();

  const int wave = threadIdx.x >> 5;
  const int lane = threadIdx.x & 31;
  const int hf   = lane >> 4;
  const int nl   = lane & 15;

  const v16h* pW0 = (const v16h*)aW0;
  const v16h* pW1 = (const v16h*)aW1;
  const v16h* pW2 = (const v16h*)aW2;

  const int totTiles = (BV * NZV) / 16;               // 131072
  const int nWaves = gridDim.x * 8;
  int gw = blockIdx.x * 8 + wave;

  for (int tile = gw; tile < totTiles; tile += nWaves) {
    int b = tile >> 14;                 // NZ/16 = 16384 tiles per batch row
    int z = (tile & 16383) * 16;

    // build scalars and layer-1 B fragment (K=0..13 live in lanes 0..15)
    v16h bv = {};
    if (hf == 0) {
      size_t pbase = (size_t)b * NZV + (size_t)z + nl;
      float ev[4], od[4];
#pragma unroll
      for (int c = 0; c < 4; ++c) ev[c] = conv[(size_t)(c * 8) * NZV + pbase].x;
#pragma unroll
      for (int c = 0; c < 4; ++c) od[c] = conv[(size_t)((c + 4) * 8) * NZV + pbase].x;
      float s[NIRRED] = { ev[0], ev[1], ev[2], ev[3],
                          od[0]*od[0], od[0]*od[1], od[0]*od[2], od[0]*od[3],
                          od[1]*od[1], od[1]*od[2], od[1]*od[3],
                          od[2]*od[2], od[2]*od[3],
                          od[3]*od[3] };
#pragma unroll
      for (int c = 0; c < NIRRED; ++c) {
        sc[wave][nl * 16 + c] = s[c];
        bv[c] = (_Float16)s[c];
      }
    }

    // layer 1: h1(64x16) = W0(64x14) @ s
    v8f h1a[4];
#pragma unroll
    for (int t = 0; t < 4; ++t) {
      v8f zz = {};
      h1a[t] = __builtin_amdgcn_wmma_f32_16x16x32_f16(false, pW0[t * 32 + lane],
                                                      false, bv, (short)0, zz, false, false);
    }
#pragma unroll
    for (int t = 0; t < 4; ++t) {
#pragma unroll
      for (int v = 0; v < 8; ++v) {
        int m = t * 16 + v + hf * 8;
        float x = h1a[t][v] + sb0[m];
        x = x > 0.0f ? x : (__expf(x) - 1.0f);
        hb[wave][nl * 64 + m] = (_Float16)x;
      }
    }

    // layer 2: h2(64x16) = W1(64x64) @ h1
    v8f d4[4];
#pragma unroll
    for (int t = 0; t < 4; ++t) { v8f zz = {}; d4[t] = zz; }
#pragma unroll
    for (int kc = 0; kc < 2; ++kc) {
      v16h bb = *(const v16h*)&hb[wave][nl * 64 + hf * 16 + kc * 32];
#pragma unroll
      for (int t = 0; t < 4; ++t)
        d4[t] = __builtin_amdgcn_wmma_f32_16x16x32_f16(false, pW1[(kc * 4 + t) * 32 + lane],
                                                       false, bb, (short)0, d4[t], false, false);
    }
#pragma unroll
    for (int t = 0; t < 4; ++t) {
#pragma unroll
      for (int v = 0; v < 8; ++v) {
        int m = t * 16 + v + hf * 8;
        float x = d4[t][v] + sb1[m];
        x = x > 0.0f ? x : (__expf(x) - 1.0f);
        hb[wave][nl * 64 + m] = (_Float16)x;
      }
    }

    // layer 3: f_out(14x16) = W2(14x64) @ h2
    v8f e8 = {};
#pragma unroll
    for (int kc = 0; kc < 2; ++kc) {
      v16h bb = *(const v16h*)&hb[wave][nl * 64 + hf * 16 + kc * 32];
      e8 = __builtin_amdgcn_wmma_f32_16x16x32_f16(false, pW2[kc * 32 + lane],
                                                  false, bb, (short)0, e8, false, false);
    }

    // weighted dot: sum_c s[c][n] * (f_out[c][n] + b2[c]) over this tile
    float part = 0.0f;
#pragma unroll
    for (int v = 0; v < 8; ++v) {
      int m = v + hf * 8;
      if (m < NIRRED) part += sc[wave][nl * 16 + m] * (e8[v] + sb2[m]);
    }
#pragma unroll
    for (int off = 16; off >= 1; off >>= 1) part += __shfl_xor(part, off, 32);
    if (lane == 0) partials[tile] = part;
  }
}

// 7) Deterministic per-batch reduction: out[b] = DZ * sum(partials of batch b)
__global__ void __launch_bounds__(256) reduce_kernel(const float* __restrict__ partials,
                                                     float* __restrict__ out) {
  const int TPB = NZV / 16;
  int b = blockIdx.x;
  const float* p = partials + (size_t)b * TPB;
  float acc = 0.0f;
  for (int i = threadIdx.x; i < TPB; i += 256) acc += p[i];
  __shared__ float red[256];
  red[threadIdx.x] = acc;
  __syncthreads();
  for (int s = 128; s > 0; s >>= 1) {
    if (threadIdx.x < s) red[threadIdx.x] += red[threadIdx.x + s];
    __syncthreads();
  }
  if (threadIdx.x == 0) out[b] = DZC * red[0];
}

// Workspace layout (bytes):
//   [0, 128MB)       bufA  (64 rows x NZ complex, FFT ping)
//   [128MB, 256MB)   bufB  (FFT pong / conv result)
//   [256MB, +8MB)    w_tilde (8 x NZH complex)
//   next 14336       packed WMMA weight fragments
//   next 512KB       per-tile partials
extern "C" void kernel_launch(void* const* d_in, const int* in_sizes, int n_in,
                              void* d_out, int out_size, void* d_ws, size_t ws_size,
                              hipStream_t stream) {
  (void)in_sizes; (void)n_in; (void)out_size; (void)ws_size;
  const float* nin = (const float*)d_in[0];
  const float* wW0 = (const float*)d_in[1];
  const float* wb0 = (const float*)d_in[2];
  const float* wW1 = (const float*)d_in[3];
  const float* wb1 = (const float*)d_in[4];
  const float* wW2 = (const float*)d_in[5];
  const float* wb2 = (const float*)d_in[6];
  const float* fW0 = (const float*)d_in[7];
  const float* fb0 = (const float*)d_in[8];
  const float* fW1 = (const float*)d_in[9];
  const float* fb1 = (const float*)d_in[10];
  const float* fW2 = (const float*)d_in[11];
  const float* fb2 = (const float*)d_in[12];
  float* out = (float*)d_out;

  const size_t BUF_BYTES = (size_t)64 * NZV * sizeof(float2);   // 128 MiB
  const size_t WT_BYTES  = (size_t)8 * NZHV * sizeof(float2);   // ~8 MiB
  char* ws = (char*)d_ws;
  float2*   bufA = (float2*)(ws);
  float2*   bufB = (float2*)(ws + BUF_BYTES);
  float2*   wt   = (float2*)(ws + 2 * BUF_BYTES);
  _Float16* aW0  = (_Float16*)(ws + 2 * BUF_BYTES + WT_BYTES);
  _Float16* aW1  = aW0 + 4 * 32 * 16;
  _Float16* aW2  = aW1 + 2 * 4 * 32 * 16;
  float* partials = (float*)((char*)(aW2 + 2 * 32 * 16));

  wfilter_kernel<<<(NZHV + 255) / 256, 256, 0, stream>>>(wW0, wb0, wW1, wb1, wW2, wb2, wt);
  prep_weights_kernel<<<1, 256, 0, stream>>>(fW0, fW1, fW2, aW0, aW1, aW2);
  loadreal_kernel<<<(BV * NZV) / 256, 256, 0, stream>>>(nin, bufA);

  // Forward FFT: 8 rows, 18 radix-2 Stockham passes, ends back in bufA.
  {
    const int total = BV * HALFNZ;
    float2 *src = bufA, *dst = bufB;
    for (int i = 0; i < LOG2NZ; ++i) {
      fft_pass_kernel<<<total / 256, 256, 0, stream>>>(src, dst, i, NZV >> i, -1.0f, total);
      float2* tmp = src; src = dst; dst = tmp;
    }
  }

  // Spectral multiply (bufA -> bufB, 64 rows), includes 1/N for the inverse.
  specmul_kernel<<<(64 * NZV) / 256, 256, 0, stream>>>(bufA, wt, bufB);

  // Inverse FFT: 64 rows, 18 passes, ends back in bufB (real conv in .x).
  {
    const int total = 64 * HALFNZ;
    float2 *src = bufB, *dst = bufA;
    for (int i = 0; i < LOG2NZ; ++i) {
      fft_pass_kernel<<<total / 256, 256, 0, stream>>>(src, dst, i, NZV >> i, 1.0f, total);
      float2* tmp = src; src = dst; dst = tmp;
    }
  }

  fmlp_kernel<<<1024, 256, 0, stream>>>(bufB, aW0, aW1, aW2, fb0, fb1, fb2, partials);
  reduce_kernel<<<BV, 256, 0, stream>>>(partials, out);
}
